// ConvAttentionHid_5755256177186
// MI455X (gfx1250) — matrix operand order repaired and checked by reading, not compiled, not run on hardware
//
#include <hip/hip_runtime.h>
#include <hip/hip_bf16.h>
#include <math.h>

// ---------------- problem constants ----------------
constexpr int Bsz = 32;     // batch
constexpr int Lc  = 4096;   // sequence length
constexpr int Cc  = 512;    // SRC == DIM1 == TRG
constexpr int Kw  = 3;      // conv kernel width

// fused kernel tiling
constexpr int TILE    = 64;         // stage-2 (L2) output columns per workgroup
constexpr int S1NT    = 5;          // stage-1 N-tiles (80 cols computed, 66 kept)
constexpr int S2NT    = 4;          // stage-2 N-tiles (64 cols)
constexpr int CTXCOLS = TILE + 18;  // 82 ctx columns (halo 2 each side + stage-1 overshoot)
constexpr int L1COLS  = TILE + 2;   // 66 L1 columns needed by conv2
constexpr int CTP     = 520;        // LDS row stride (bf16) for transposed tiles:
constexpr int LTP     = 520;        //   520*2B = 1040B -> 4-bank col step, 16B aligned
constexpr int KB      = Cc / 32;    // 16 k-steps of 32 per tap

// ---------------- WMMA types ----------------
typedef __bf16 v16bf __attribute__((ext_vector_type(16)));
typedef __bf16 v8bf  __attribute__((ext_vector_type(8)));
typedef float  v8f   __attribute__((ext_vector_type(8)));

__device__ __forceinline__ unsigned short bf_bits(float f) {
  __bf16 h = (__bf16)f;                      // hardware f32->bf16 convert
  return __builtin_bit_cast(unsigned short, h);
}
__device__ __forceinline__ float bf2f(unsigned short s) {
  union { unsigned u; float f; } x; x.u = ((unsigned)s) << 16;
  return x.f;
}
__device__ __forceinline__ v16bf cat8(v8bf lo, v8bf hi) {
  return __builtin_shufflevector(lo, hi, 0,1,2,3,4,5,6,7,8,9,10,11,12,13,14,15);
}

// =====================================================================
// Kernel 0: repack conv weights [C][C][3] f32 -> [3][C][C] bf16
// (one-time, 0.75M elems per weight; makes A-fragments two b128 loads)
// =====================================================================
__global__ __launch_bounds__(256)
void kpack(const float* __restrict__ w, unsigned short* __restrict__ wp)
{
  const int i  = blockIdx.x * 256 + threadIdx.x;   // [kk][co][ci] linear
  const int ci = i & (Cc - 1);
  const int t  = i >> 9;
  const int co = t & (Cc - 1);
  const int kk = t >> 9;
  wp[i] = bf_bits(w[((size_t)co * Cc + ci) * Kw + kk]);
}

// =====================================================================
// Kernel 1: fused conv1 + bias + relu + conv2 + bias + gate, L2 -> bf16
// One 512-thread workgroup per (l-tile, batch); 16 waves, wave w owns
// output rows co in [32w, 32w+32) (two 16-row M-tiles), all N-tiles.
// LDS tiles are TRANSPOSED ([col][channel]) so B-fragments are two
// contiguous ds_load_b128 per lane; weights pre-packed so A-fragments
// are two contiguous bf16 global_load_b128 per lane.
// =====================================================================
__global__ __launch_bounds__(512)
void fused_conv12(const float* __restrict__ context,
                  const unsigned short* __restrict__ w1p,
                  const float* __restrict__ b1,
                  const unsigned short* __restrict__ w2p,
                  const float* __restrict__ b2,
                  const float* __restrict__ gate,
                  unsigned short* __restrict__ L2out)
{
  __shared__ unsigned short s_ctxT[CTXCOLS * CTP];  // 85280 B  [col][ci]
  __shared__ unsigned short s_l1T [L1COLS  * LTP];  // 68640 B  [col][co]
  __shared__ float s_b1v[Cc];
  __shared__ float s_b2v[Cc];
  __shared__ float s_gv [Cc];

  const int b    = blockIdx.y;
  const int l0   = blockIdx.x * TILE;
  const int tid  = threadIdx.x;
  const int lane = tid & 31;
  const int wave = tid >> 5;            // 0..15
  const int n16  = lane & 15;           // N col (B/C) or M row (A) in fragment
  const int kg   = lane >> 4;           // K-half select
  const int mb0  = wave * 32;           // first co row owned by this wave

  // ---- stage 0: biases/gate + transposed ctx tile into LDS (bf16) ----
  s_b1v[tid] = b1[tid];
  s_b2v[tid] = b2[tid];
  s_gv [tid] = gate[b * Cc + tid];
  {
    const int jsub = tid >> 8;            // 0..1
    const int c2   = (tid & 255) * 2;     // even channel pair
    for (int j = jsub; j < CTXCOLS; j += 2) {
      const int l = l0 - 2 + j;
      float2 v = make_float2(0.0f, 0.0f);
      if (l >= 0 && l < Lc)
        v = *(const float2*)&context[((size_t)b * Lc + l) * Cc + c2];
      const unsigned pair = (unsigned)bf_bits(v.x) | ((unsigned)bf_bits(v.y) << 16);
      *(unsigned*)&s_ctxT[j * CTP + c2] = pair;
    }
  }
  __syncthreads();

  // ---- stage 1: L1 = relu(conv1(ctx) + b1), 80 cols computed, 66 kept ----
  {
    v8f acc[2][S1NT];
#pragma unroll
    for (int mt = 0; mt < 2; ++mt)
#pragma unroll
      for (int nt = 0; nt < S1NT; ++nt) acc[mt][nt] = (v8f){0,0,0,0,0,0,0,0};

#pragma unroll 1
    for (int kk = 0; kk < Kw; ++kk) {
      // A row base (weights, bf16): elements e<8 at ci=k0+8kg, e>=8 at +16
      const unsigned short* a0 = w1p + ((size_t)kk * Cc + mb0 + n16) * Cc + 8 * kg;
      const unsigned short* a1 = a0 + (size_t)16 * Cc;
#pragma unroll 1
      for (int kb = 0; kb < KB; ++kb) {
        const int k0 = kb * 32;
        v16bf a[2];
        a[0] = cat8(*(const v8bf*)(a0 + k0), *(const v8bf*)(a0 + k0 + 16));
        a[1] = cat8(*(const v8bf*)(a1 + k0), *(const v8bf*)(a1 + k0 + 16));
#pragma unroll
        for (int nt = 0; nt < S1NT; ++nt) {
          const unsigned short* bp = &s_ctxT[(nt * 16 + n16 + kk) * CTP + k0 + 8 * kg];
          const v16bf bv = cat8(*(const v8bf*)(bp), *(const v8bf*)(bp + 16));
          acc[0][nt] = __builtin_amdgcn_wmma_f32_16x16x32_bf16(
              false, a[0], false, bv, (short)0, acc[0][nt], false, false);
          acc[1][nt] = __builtin_amdgcn_wmma_f32_16x16x32_bf16(
              false, a[1], false, bv, (short)0, acc[1][nt], false, false);
        }
      }
    }

    // epilogue: bias + relu, transposed store: 8 consecutive co = 1 b128
    const int hh = kg;
#pragma unroll
    for (int mt = 0; mt < 2; ++mt) {
      const int cobase = mb0 + mt * 16 + 8 * hh;
      float bias1[8];
#pragma unroll
      for (int v = 0; v < 8; ++v) bias1[v] = s_b1v[cobase + v];
#pragma unroll
      for (int nt = 0; nt < S1NT; ++nt) {
        const int j1 = nt * 16 + n16;
        if (j1 < L1COLS) {
          const int l = l0 - 1 + j1;
          const float keep = ((unsigned)l < (unsigned)Lc) ? 1.0f : 0.0f;
          v8bf pk;
#pragma unroll
          for (int v = 0; v < 8; ++v)
            pk[v] = (__bf16)(fmaxf(acc[mt][nt][v] + bias1[v], 0.0f) * keep);
          *(v8bf*)&s_l1T[j1 * LTP + cobase] = pk;
        }
      }
    }
  }
  __syncthreads();

  // ---- stage 2: L2 = (conv2(L1) + b2) * gate over 64 cols ----
  {
    v8f acc[2][S2NT];
#pragma unroll
    for (int mt = 0; mt < 2; ++mt)
#pragma unroll
      for (int nt = 0; nt < S2NT; ++nt) acc[mt][nt] = (v8f){0,0,0,0,0,0,0,0};

#pragma unroll 1
    for (int kk = 0; kk < Kw; ++kk) {
      const unsigned short* a0 = w2p + ((size_t)kk * Cc + mb0 + n16) * Cc + 8 * kg;
      const unsigned short* a1 = a0 + (size_t)16 * Cc;
#pragma unroll 1
      for (int kb = 0; kb < KB; ++kb) {
        const int k0 = kb * 32;
        v16bf a[2];
        a[0] = cat8(*(const v8bf*)(a0 + k0), *(const v8bf*)(a0 + k0 + 16));
        a[1] = cat8(*(const v8bf*)(a1 + k0), *(const v8bf*)(a1 + k0 + 16));
#pragma unroll
        for (int nt = 0; nt < S2NT; ++nt) {
          const unsigned short* bp = &s_l1T[(nt * 16 + n16 + kk) * LTP + k0 + 8 * kg];
          const v16bf bv = cat8(*(const v8bf*)(bp), *(const v8bf*)(bp + 16));
          acc[0][nt] = __builtin_amdgcn_wmma_f32_16x16x32_bf16(
              false, a[0], false, bv, (short)0, acc[0][nt], false, false);
          acc[1][nt] = __builtin_amdgcn_wmma_f32_16x16x32_bf16(
              false, a[1], false, bv, (short)0, acc[1][nt], false, false);
        }
      }
    }

    // epilogue: bias, gate, store bf16 rows of L2 to workspace
    const int hh = kg;
#pragma unroll
    for (int mt = 0; mt < 2; ++mt) {
      const int cobase = mb0 + mt * 16 + 8 * hh;
      float bias2[8], gg[8];
#pragma unroll
      for (int v = 0; v < 8; ++v) { bias2[v] = s_b2v[cobase + v]; gg[v] = s_gv[cobase + v]; }
#pragma unroll
      for (int v = 0; v < 8; ++v) {
        unsigned short* dst = L2out + ((size_t)b * Cc + cobase + v) * Lc + l0 + n16;
#pragma unroll
        for (int nt = 0; nt < S2NT; ++nt) {
          const float x = (acc[mt][nt][v] + bias2[v]) * gg[v];
          dst[nt * 16] = bf_bits(x);
        }
      }
    }
  }
}

// =====================================================================
// Kernel 2a: nrm[b*C+co] = || L2[b,co,:] ||_2
// =====================================================================
__global__ __launch_bounds__(256)
void knorm(const unsigned short* __restrict__ L2, float* __restrict__ nrm)
{
  __shared__ float red[256];
  const int row = blockIdx.x;                  // b*C + co
  const int t = threadIdx.x;
  const unsigned short* p = L2 + (size_t)row * Lc;
  float s = 0.0f;
  for (int l = t; l < Lc; l += 256) { float x = bf2f(p[l]); s += x * x; }
  red[t] = s; __syncthreads();
  for (int w = 128; w > 0; w >>= 1) { if (t < w) red[t] += red[t + w]; __syncthreads(); }
  if (t == 0) nrm[row] = sqrtf(red[0]);
}

// =====================================================================
// Kernel 2b: attn[b,l] = b3 + sum_co sum_k (w3[co,k]/nrm[b,co]) * L2[b,co,l+k-1]
// =====================================================================
__global__ __launch_bounds__(256)
void kattn(const unsigned short* __restrict__ L2, const float* __restrict__ nrm,
           const float* __restrict__ w3, const float* __restrict__ b3,
           float* __restrict__ attn)
{
  __shared__ float sc[3 * Cc];
  const int b = blockIdx.y;
  const int l = blockIdx.x * 256 + threadIdx.x;
  for (int i = threadIdx.x; i < Cc; i += 256) {
    const float inv = 1.0f / nrm[b * Cc + i];
    sc[i]          = w3[i * 3 + 0] * inv;
    sc[Cc + i]     = w3[i * 3 + 1] * inv;
    sc[2 * Cc + i] = w3[i * 3 + 2] * inv;
  }
  __syncthreads();
  float acc = b3[0];
  const unsigned short* base = L2 + ((size_t)b * Cc) * Lc + l;
#pragma unroll 4
  for (int co = 0; co < Cc; ++co) {
    const unsigned short* p = base + (size_t)co * Lc;
    const float xm = (l > 0)      ? bf2f(p[-1]) : 0.0f;
    const float x0 =                bf2f(p[0]);
    const float xp = (l < Lc - 1) ? bf2f(p[1]) : 0.0f;
    acc += sc[co] * xm + sc[Cc + co] * x0 + sc[2 * Cc + co] * xp;
  }
  attn[(size_t)b * Lc + l] = acc;
}

// =====================================================================
// Kernel 3: softmax over L per batch (attn -> sm in workspace)
// =====================================================================
__global__ __launch_bounds__(256)
void ksoftmax(const float* __restrict__ attn, float* __restrict__ sm)
{
  __shared__ float red[256];
  __shared__ float s_bc;
  const int b = blockIdx.x, t = threadIdx.x;
  const float* a = attn + (size_t)b * Lc;
  float m = -3.402823466e38f;
  for (int l = t; l < Lc; l += 256) m = fmaxf(m, a[l]);
  red[t] = m; __syncthreads();
  for (int w = 128; w > 0; w >>= 1) { if (t < w) red[t] = fmaxf(red[t], red[t + w]); __syncthreads(); }
  if (t == 0) s_bc = red[0];
  __syncthreads();
  const float mx = s_bc;
  __syncthreads();
  float s = 0.0f;
  for (int l = t; l < Lc; l += 256) s += __expf(a[l] - mx);
  red[t] = s; __syncthreads();
  for (int w = 128; w > 0; w >>= 1) { if (t < w) red[t] += red[t + w]; __syncthreads(); }
  if (t == 0) s_bc = red[0];
  __syncthreads();
  const float inv = 1.0f / s_bc;
  for (int l = t; l < Lc; l += 256) sm[(size_t)b * Lc + l] = __expf(a[l] - mx) * inv;
}

// =====================================================================
// Kernel 4: weighted_context[b,c] = sum_l context[b,l,c] * sm[b,l]
// =====================================================================
__global__ __launch_bounds__(512)
void kwctx(const float* __restrict__ context, const float* __restrict__ sm,
           float* __restrict__ wc)
{
  __shared__ float w[Lc];           // 16 KB
  const int b = blockIdx.x, c = threadIdx.x;
  for (int l = c; l < Lc; l += 512) w[l] = sm[(size_t)b * Lc + l];
  __syncthreads();
  const float* p = context + (size_t)b * Lc * Cc + c;
  float acc = 0.0f;
#pragma unroll 8
  for (int l = 0; l < Lc; ++l) acc += p[(size_t)l * Cc] * w[l];
  wc[b * Cc + c] = acc;
}

// =====================================================================
// Kernel 5: h_tilde = tanh([wc, input] @ W_out^T)    [32,1024] x [1024,512]
// =====================================================================
__global__ __launch_bounds__(512)
void kout(const float* __restrict__ wc, const float* __restrict__ inp,
          const float* __restrict__ Wo, float* __restrict__ out_h)
{
  __shared__ float hc[2 * Cc];      // 4 KB
  const int b = blockIdx.x, t = threadIdx.x;
  hc[t]      = wc[b * Cc + t];
  hc[Cc + t] = inp[b * Cc + t];
  __syncthreads();
  const float* wr = Wo + (size_t)t * (2 * Cc);
  float acc = 0.0f;
#pragma unroll 8
  for (int j = 0; j < 2 * Cc; ++j) acc += hc[j] * wr[j];
  out_h[b * Cc + t] = tanhf(acc);
}

// =====================================================================
extern "C" void kernel_launch(void* const* d_in, const int* in_sizes, int n_in,
                              void* d_out, int out_size, void* d_ws, size_t ws_size,
                              hipStream_t stream) {
  (void)in_sizes; (void)n_in; (void)out_size; (void)ws_size;
  const float* input   = (const float*)d_in[0];   // [B, TRG]
  const float* context = (const float*)d_in[1];   // [B, L, SRC]
  // d_in[2] = src_emb (unused by the reference)
  const float* w1 = (const float*)d_in[3];
  const float* b1 = (const float*)d_in[4];
  const float* w2 = (const float*)d_in[5];
  const float* b2 = (const float*)d_in[6];
  const float* w3 = (const float*)d_in[7];
  const float* b3 = (const float*)d_in[8];
  const float* Wo = (const float*)d_in[9];

  float* out_h    = (float*)d_out;                 // [B*TRG]
  float* out_attn = (float*)d_out + Bsz * Cc;      // [B*L]

  char* ws = (char*)d_ws;
  const size_t szL2  = (size_t)Bsz * Cc * Lc * 2;  // 128 MB bf16
  unsigned short* L2  = (unsigned short*)ws;
  float* nrm = (float*)(ws + szL2);                           // 64 KB
  float* smx = nrm + Bsz * Cc;                                // 512 KB
  float* wc  = smx + Bsz * Lc;                                // 64 KB
  unsigned short* w1p = (unsigned short*)(wc + Bsz * Cc);     // 1.5 MB bf16
  unsigned short* w2p = w1p + (size_t)Kw * Cc * Cc;           // 1.5 MB bf16

  kpack<<<dim3(Kw * Cc * Cc / 256), 256, 0, stream>>>(w1, w1p);
  kpack<<<dim3(Kw * Cc * Cc / 256), 256, 0, stream>>>(w2, w2p);
  fused_conv12<<<dim3(Lc / TILE, Bsz), 512, 0, stream>>>(context, w1p, b1, w2p, b2, input, L2);
  knorm    <<<dim3(Bsz * Cc), 256, 0, stream>>>(L2, nrm);
  kattn    <<<dim3(Lc / 256, Bsz), 256, 0, stream>>>(L2, nrm, w3, b3, out_attn);
  ksoftmax <<<dim3(Bsz), 256, 0, stream>>>(out_attn, smx);
  kwctx    <<<dim3(Bsz), 512, 0, stream>>>(context, smx, wc);
  kout     <<<dim3(Bsz), 512, 0, stream>>>(wc, input, Wo, out_h);
}
